// NerModel_29815662969187
// MI455X (gfx1250) — matrix operand order repaired and checked by reading, not compiled, not run on hardware
//
#include <hip/hip_runtime.h>
#include <hip/hip_bf16.h>

// ---------------------------------------------------------------------------
// Model dims (hardcoded from reference)
// ---------------------------------------------------------------------------
#define Bz   256
#define Sz   64
#define Hz   768
#define Lz   12
#define NHz  12
#define DHz  64
#define Fz   3072
#define Tz   11
#define NBS  (Bz * Sz)        // 16384 token rows
#define QKVN (3 * Hz)         // 2304

typedef __attribute__((ext_vector_type(16))) __bf16 bf16x16;
typedef __attribute__((ext_vector_type(8)))  float  f32x8;

// ---------------------------------------------------------------------------
// CDNA5 async global->LDS staging (GLOBAL_LOAD_ASYNC_TO_LDS_B128, ASYNCcnt)
// GVS addressing: mem = SGPR_base64 + VGPR_offset32 + inst_offset
// LDS dest address = low 32 bits of generic shared pointer (ISA aperture map)
// ---------------------------------------------------------------------------
__device__ __forceinline__ unsigned lds_off(const void* p) {
  return (unsigned)(unsigned long long)p;
}

__device__ __forceinline__ void async_load_b128(unsigned lds_addr, unsigned gl_byte_off,
                                                unsigned long long sbase) {
  asm volatile("global_load_async_to_lds_b128 %0, %1, %2"
               :: "v"(lds_addr), "v"(gl_byte_off), "s"(sbase)
               : "memory");
}

__device__ __forceinline__ void wait_async0() {
  asm volatile("s_wait_asynccnt 0x0" ::: "memory");
}

// ---------------------------------------------------------------------------
// WMMA fragment loaders (contiguous k = 16*(lane/16) + e in both A and B, wave32)
// ---------------------------------------------------------------------------

// A fragment 16x32 bf16 from row-major array (lda in elements).
// lanes 0-15: rows M=lane, K = 0..15; lanes 16-31: K = 16..31
__device__ inline bf16x16 load_a_frag(const __bf16* A, int lda, int row0, int k0, int lane) {
  const int r  = row0 + (lane & 15);
  const int kb = k0 + ((lane >> 4) << 4);
  const __bf16* p = A + r * lda + kb;
  bf16x16 v;
#pragma unroll
  for (int e = 0; e < 8; ++e) v[e] = p[e];
#pragma unroll
  for (int e = 0; e < 8; ++e) v[8 + e] = p[8 + e];
  return v;
}

// B fragment 32x16 where logical B[k][n] = M[n][k] (M is [N][K] row-major).
// lane = column n; lanes 0-15 hold K=0..15, lanes 16-31 hold K=16..31.
// Reads contiguous 16 bf16 per lane -> vectorizes to 2 x ds_load_b128.
__device__ inline bf16x16 load_bt_frag(const __bf16* M, int ldm, int k0, int n0, int lane) {
  const int n  = n0 + (lane & 15);
  const int kb = k0 + ((lane >> 4) << 4);
  const __bf16* p = M + n * ldm + kb;
  bf16x16 v;
#pragma unroll
  for (int e = 0; e < 16; ++e) v[e] = p[e];
  return v;
}

__device__ inline f32x8 wmma_bf16(bf16x16 a, bf16x16 b, f32x8 c) {
  return __builtin_amdgcn_wmma_f32_16x16x32_bf16(false, a, false, b, (short)0, c, false, false);
}

__device__ inline float gelu_tanh(float x) {
  const float c0 = 0.7978845608028654f;   // sqrt(2/pi)
  float inner = c0 * (x + 0.044715f * x * x * x);
  return 0.5f * x * (1.0f + tanhf(inner));
}

// ---------------------------------------------------------------------------
// bf16 GEMM with TRANSPOSED weights: C[M,N] = act(A[M,K] @ Bt[N,K]^T + bias)
// Block: 256 threads (8 waves). Tile 128x64, BK=32, double-buffered LDS fed by
// GLOBAL_LOAD_ASYNC_TO_LDS_B128. Each wave: 4 WMMA tiles (4 rows x 1 col).
// All LDS fragment reads are contiguous 16B vectors.
// ACT: 0 = identity, 1 = gelu(tanh)
// ---------------------------------------------------------------------------
#define BM 128
#define BN 64
#define BK 32

template <int ACT>
__global__ __launch_bounds__(256) void gemm_bf16_kernel(
    const __bf16* __restrict__ A, const __bf16* __restrict__ Bt,
    const float* __restrict__ bias,
    float* __restrict__ Cf, __bf16* __restrict__ Cb,
    int M, int N, int K) {
  __shared__ __align__(16) __bf16 As[2][BM * BK];    // 2 x 8KB  (row-major [m][k])
  __shared__ __align__(16) __bf16 BsT[2][BN * BK];   // 2 x 4KB  (row-major [n][k])

  const int  tid  = threadIdx.x;
  const int  lane = tid & 31;
  const int  wave = tid >> 5;
  const long m0   = (long)blockIdx.y * BM;
  const long n0   = (long)blockIdx.x * BN;
  const int  wr   = wave >> 2;         // 0..1 -> tile rows 4*wr .. 4*wr+3
  const int  wc   = wave & 3;          // tile col 0..3

  // staging coords: A 128x32 -> 16 el/thread (2 x b128); Bt 64x32 -> 8 el (1 x b128)
  const int ar = tid >> 1, ac = (tid & 1) << 4;
  const int bn = tid >> 2, bk = (tid & 3) << 3;

  const unsigned long long Abase = (unsigned long long)A;
  const unsigned long long Bbase = (unsigned long long)Bt;

  f32x8 acc[4] = {f32x8{}, f32x8{}, f32x8{}, f32x8{}};

  auto stage = [&](int buf, int k0) {
    unsigned la = lds_off(&As[buf][ar * BK + ac]);
    unsigned ga = (unsigned)(((m0 + ar) * (long)K + k0 + ac) * 2);
    async_load_b128(la, ga, Abase);
    async_load_b128(la + 16u, ga + 16u, Abase);
    unsigned lb = lds_off(&BsT[buf][bn * BK + bk]);
    unsigned gb = (unsigned)(((n0 + bn) * (long)K + k0 + bk) * 2);
    async_load_b128(lb, gb, Bbase);
  };

  stage(0, 0);
  wait_async0();
  __syncthreads();

  int cur = 0;
  for (int k0 = 0; k0 < K; k0 += BK) {
    if (k0 + BK < K) stage(cur ^ 1, k0 + BK);

    bf16x16 bf = load_bt_frag(&BsT[cur][0], BK, 0, wc * 16, lane);
#pragma unroll
    for (int t = 0; t < 4; ++t) {
      bf16x16 af = load_a_frag(&As[cur][0], BK, (4 * wr + t) * 16, 0, lane);
      acc[t] = wmma_bf16(af, bf, acc[t]);
    }

    wait_async0();
    __syncthreads();
    cur ^= 1;
  }

  // epilogue: lane -> column; VGPR r -> row r + 8*(lane>=16)
  const long n   = n0 + wc * 16 + (lane & 15);
  const float bv = bias ? bias[n] : 0.0f;
  const int rofs = (lane >> 4) << 3;
#pragma unroll
  for (int t = 0; t < 4; ++t) {
#pragma unroll
    for (int r = 0; r < 8; ++r) {
      long m = m0 + (4 * wr + t) * 16 + r + rofs;
      float v = acc[t][r] + bv;
      if (ACT == 1) v = gelu_tanh(v);
      long off = m * (long)N + n;
      if (Cf) Cf[off] = v;
      if (Cb) Cb[off] = (__bf16)v;
    }
  }
}

// ---------------------------------------------------------------------------
// Fused attention per (b, h): S = softmax(Q K^T / 8 + maskbias); O = S V
// Q,K staged async; V staged transposed (Vt[d][t]) so P.V B-fragments are
// contiguous; P (probs) reuses Ks.
// ---------------------------------------------------------------------------
__global__ __launch_bounds__(256) void attention_kernel(
    const __bf16* __restrict__ qkv,      // [NBS, 2304]
    const int* __restrict__ mask,        // [B, S]
    __bf16* __restrict__ out) {          // [NBS, 768]
  __shared__ __align__(16) __bf16 Qs[Sz * DHz];
  __shared__ __align__(16) __bf16 Ks[Sz * DHz];   // reused as P after softmax
  __shared__ __align__(16) __bf16 Vt[DHz * Sz];   // transposed V: Vt[d][t]
  __shared__ float Ss[Sz * Sz];

  const int bh = blockIdx.x;
  const int b  = bh / NHz;
  const int h  = bh - b * NHz;
  const int tid  = threadIdx.x;
  const int lane = tid & 31;
  const int wave = tid >> 5;

  const int r  = tid >> 2;          // 0..63 row (s or t index)
  const int c0 = (tid & 3) << 4;    // 0,16,32,48 channel chunk

  // async staging of Q,K
  {
    const unsigned long long base = (unsigned long long)qkv;
    const unsigned gq = (unsigned)((((long)(b * Sz + r)) * QKVN + h * DHz + c0) * 2);
    const unsigned lq = lds_off(&Qs[r * DHz + c0]);
    const unsigned lk = lds_off(&Ks[r * DHz + c0]);
    async_load_b128(lq,       gq,                 base);
    async_load_b128(lq + 16u, gq + 16u,           base);
    async_load_b128(lk,       gq + 2u * Hz,       base);
    async_load_b128(lk + 16u, gq + 2u * Hz + 16u, base);
  }
  // V staged transposed via registers (global reads contiguous, LDS stores strided)
  {
    const __bf16* vrow = qkv + ((long)(b * Sz + r)) * QKVN + 2 * Hz + h * DHz + c0;
#pragma unroll
    for (int j = 0; j < 16; ++j) Vt[(c0 + j) * Sz + r] = vrow[j];
  }
  wait_async0();
  __syncthreads();

  const int tr0 = (wave >> 2) * 2;
  const int tc  = (wave & 3);

  // ---- S = Q K^T ----
  {
    f32x8 acc0{};
    f32x8 acc1{};
#pragma unroll
    for (int kk = 0; kk < DHz; kk += 32) {
      bf16x16 bf = load_bt_frag(Ks, DHz, kk, tc * 16, lane);   // B[d][t] = K[t][d]
      bf16x16 a0 = load_a_frag(Qs, DHz, tr0 * 16, kk, lane);
      bf16x16 a1 = load_a_frag(Qs, DHz, (tr0 + 1) * 16, kk, lane);
      acc0 = wmma_bf16(a0, bf, acc0);
      acc1 = wmma_bf16(a1, bf, acc1);
    }
    const int col   = tc * 16 + (lane & 15);
    const float mb  = (1.0f - (float)mask[b * Sz + col]) * -1e9f;
    const int rofs  = (lane >> 4) << 3;
#pragma unroll
    for (int rr = 0; rr < 8; ++rr) {
      Ss[(tr0 * 16 + rr + rofs) * Sz + col]       = acc0[rr] * 0.125f + mb;
      Ss[((tr0 + 1) * 16 + rr + rofs) * Sz + col] = acc1[rr] * 0.125f + mb;
    }
  }
  __syncthreads();

  // ---- softmax rows -> P (reuse Ks) ----
  if (tid < Sz) {
    float mx = -3.402823e38f;
    for (int t = 0; t < Sz; ++t) mx = fmaxf(mx, Ss[tid * Sz + t]);
    float sum = 0.0f;
    for (int t = 0; t < Sz; ++t) {
      float e = __expf(Ss[tid * Sz + t] - mx);
      Ss[tid * Sz + t] = e;
      sum += e;
    }
    float inv = 1.0f / sum;
    for (int t = 0; t < Sz; ++t) Ks[tid * Sz + t] = (__bf16)(Ss[tid * Sz + t] * inv);
  }
  __syncthreads();

  // ---- O = P V  (B[t][d] fragment = Vt[d][t], contiguous per lane) ----
  {
    f32x8 acc0{};
    f32x8 acc1{};
#pragma unroll
    for (int kk = 0; kk < Sz; kk += 32) {
      bf16x16 bf = load_bt_frag(Vt, Sz, kk, tc * 16, lane);
      bf16x16 a0 = load_a_frag(Ks, Sz, tr0 * 16, kk, lane);
      bf16x16 a1 = load_a_frag(Ks, Sz, (tr0 + 1) * 16, kk, lane);
      acc0 = wmma_bf16(a0, bf, acc0);
      acc1 = wmma_bf16(a1, bf, acc1);
    }
    const int col  = tc * 16 + (lane & 15);
    const int rofs = (lane >> 4) << 3;
#pragma unroll
    for (int rr = 0; rr < 8; ++rr) {
      {
        int row = tr0 * 16 + rr + rofs;
        out[((long)(b * Sz + row)) * Hz + h * DHz + col] = (__bf16)acc0[rr];
      }
      {
        int row = (tr0 + 1) * 16 + rr + rofs;
        out[((long)(b * Sz + row)) * Hz + h * DHz + col] = (__bf16)acc1[rr];
      }
    }
  }
}

// ---------------------------------------------------------------------------
// fp32 [K,N] -> bf16 transposed [N,K], 32x32 LDS-tiled (coalesced both sides)
// ---------------------------------------------------------------------------
__global__ __launch_bounds__(256) void cvt_transpose_kernel(
    const float* __restrict__ src, __bf16* __restrict__ dst, int K, int N) {
  __shared__ float t[32][33];
  const int k0 = blockIdx.y * 32;
  const int n0 = blockIdx.x * 32;
  const int tx = threadIdx.x & 31;
  const int ty = threadIdx.x >> 5;   // 0..7
#pragma unroll
  for (int i = 0; i < 32; i += 8)
    t[ty + i][tx] = src[(long)(k0 + ty + i) * N + n0 + tx];
  __syncthreads();
#pragma unroll
  for (int i = 0; i < 32; i += 8)
    dst[(long)(n0 + ty + i) * K + k0 + tx] = (__bf16)t[tx][ty + i];
}

// ---------------------------------------------------------------------------
// block reduction helper (256 threads)
// ---------------------------------------------------------------------------
__device__ inline float block_sum256(float v, float* red) {
  const int t = threadIdx.x;
  red[t] = v;
  __syncthreads();
  for (int s = 128; s > 0; s >>= 1) {
    if (t < s) red[t] += red[t + s];
    __syncthreads();
  }
  float r = red[0];
  __syncthreads();
  return r;
}

// ---------------------------------------------------------------------------
// embedding gather + layernorm  (one block per token row, 256 threads)
// ---------------------------------------------------------------------------
__global__ __launch_bounds__(256) void embed_ln_kernel(
    const int* __restrict__ ids, const float* __restrict__ wemb,
    const float* __restrict__ pemb, const float* __restrict__ temb,
    const float* __restrict__ g, const float* __restrict__ bb,
    float* __restrict__ x, __bf16* __restrict__ xb) {
  __shared__ float red[256];
  const long row = blockIdx.x;
  const int  s   = (int)(row & (Sz - 1));
  const long id  = ids[row];

  float vals[3];
  float sum = 0.0f;
#pragma unroll
  for (int i = 0; i < 3; ++i) {
    int hh = threadIdx.x + i * 256;
    float v = wemb[id * Hz + hh] + pemb[(long)s * Hz + hh] + temb[hh];
    vals[i] = v;
    sum += v;
  }
  float mean = block_sum256(sum, red) * (1.0f / Hz);
  float vs = 0.0f;
#pragma unroll
  for (int i = 0; i < 3; ++i) {
    float d = vals[i] - mean;
    vs += d * d;
  }
  float var  = block_sum256(vs, red) * (1.0f / Hz);
  float rstd = rsqrtf(var + 1e-12f);
#pragma unroll
  for (int i = 0; i < 3; ++i) {
    int hh = threadIdx.x + i * 256;
    float y = (vals[i] - mean) * rstd * g[hh] + bb[hh];
    x[row * Hz + hh]  = y;
    xb[row * Hz + hh] = (__bf16)y;
  }
}

// ---------------------------------------------------------------------------
// residual + layernorm (in-place on x), one block per row
// ---------------------------------------------------------------------------
__global__ __launch_bounds__(256) void residual_ln_kernel(
    float* __restrict__ x, const float* __restrict__ delta,
    const float* __restrict__ g, const float* __restrict__ bb,
    __bf16* __restrict__ xb) {
  __shared__ float red[256];
  const long row = blockIdx.x;
  float vals[3];
  float sum = 0.0f;
#pragma unroll
  for (int i = 0; i < 3; ++i) {
    int hh = threadIdx.x + i * 256;
    float v = x[row * Hz + hh] + delta[row * Hz + hh];
    vals[i] = v;
    sum += v;
  }
  float mean = block_sum256(sum, red) * (1.0f / Hz);
  float vs = 0.0f;
#pragma unroll
  for (int i = 0; i < 3; ++i) {
    float d = vals[i] - mean;
    vs += d * d;
  }
  float var  = block_sum256(vs, red) * (1.0f / Hz);
  float rstd = rsqrtf(var + 1e-12f);
#pragma unroll
  for (int i = 0; i < 3; ++i) {
    int hh = threadIdx.x + i * 256;
    float y = (vals[i] - mean) * rstd * g[hh] + bb[hh];
    x[row * Hz + hh]  = y;
    xb[row * Hz + hh] = (__bf16)y;
  }
}

// ---------------------------------------------------------------------------
// logits: [NBS,768] @ [768,11] + bias   (one block per row)
// ---------------------------------------------------------------------------
__global__ __launch_bounds__(64) void logits_kernel(
    const float* __restrict__ x, const float* __restrict__ fcw,
    const float* __restrict__ fcb, float* __restrict__ logits) {
  __shared__ float xr[Hz];
  const long row = blockIdx.x;
  for (int i = threadIdx.x; i < Hz; i += 64) xr[i] = x[row * Hz + i];
  __syncthreads();
  if (threadIdx.x < Tz) {
    float s = fcb[threadIdx.x];
    for (int k = 0; k < Hz; ++k) s += xr[k] * fcw[k * Tz + threadIdx.x];
    logits[row * Tz + threadIdx.x] = s;
  }
}

// ---------------------------------------------------------------------------
// CRF log-likelihood per batch element (alpha recursion over T=11 tags)
// ---------------------------------------------------------------------------
__global__ __launch_bounds__(32) void crf_llh_kernel(
    const float* __restrict__ em, const int* __restrict__ tags,
    const int* __restrict__ mask, const float* __restrict__ start,
    const float* __restrict__ endv, const float* __restrict__ trans,
    float* __restrict__ llh) {
  const int b = blockIdx.x;
  const float* E = em + (long)b * Sz * Tz;
  __shared__ float alpha[Tz], nalpha[Tz];
  const int t = threadIdx.x;

  if (t < Tz) alpha[t] = start[t] + E[t];
  __syncthreads();
  for (int s = 1; s < Sz; ++s) {
    float m = (float)mask[b * Sz + s];
    if (t < Tz) {
      float mx = -3.402823e38f;
      for (int i = 0; i < Tz; ++i) mx = fmaxf(mx, alpha[i] + trans[i * Tz + t]);
      float ss = 0.0f;
      for (int i = 0; i < Tz; ++i) ss += __expf(alpha[i] + trans[i * Tz + t] - mx);
      float nv = mx + __logf(ss) + E[s * Tz + t];
      nalpha[t] = (m > 0.0f) ? nv : alpha[t];
    }
    __syncthreads();
    if (t < Tz) alpha[t] = nalpha[t];
    __syncthreads();
  }
  if (t == 0) {
    float mx = -3.402823e38f;
    for (int i = 0; i < Tz; ++i) mx = fmaxf(mx, alpha[i] + endv[i]);
    float ss = 0.0f;
    for (int i = 0; i < Tz; ++i) ss += __expf(alpha[i] + endv[i] - mx);
    float denom = mx + __logf(ss);

    int prev = tags[b * Sz];
    float score = start[prev] + E[prev];
    for (int s = 1; s < Sz; ++s) {
      float m = (float)mask[b * Sz + s];
      int cur = tags[b * Sz + s];
      score += (trans[prev * Tz + cur] + E[s * Tz + cur]) * m;
      prev = cur;
    }
    int cnt = 0;
    for (int s = 0; s < Sz; ++s) cnt += mask[b * Sz + s];
    score += endv[tags[b * Sz + (cnt - 1)]];
    llh[b] = score - denom;
  }
}

__global__ __launch_bounds__(256) void loss_kernel(const float* __restrict__ llh,
                                                   float* __restrict__ out) {
  __shared__ float red[256];
  float v = llh[threadIdx.x];
  float s = block_sum256(v, red);
  if (threadIdx.x == 0) out[0] = -s * (1.0f / Bz);
}

// ---------------------------------------------------------------------------
// Viterbi decode per batch element -> pred tags (as float) into d_out[0..NBS)
// ---------------------------------------------------------------------------
__global__ __launch_bounds__(32) void viterbi_kernel(
    const float* __restrict__ em, const int* __restrict__ mask,
    const float* __restrict__ start, const float* __restrict__ endv,
    const float* __restrict__ trans, float* __restrict__ pred) {
  const int b = blockIdx.x;
  const float* E = em + (long)b * Sz * Tz;
  __shared__ float score[Tz], ns[Tz];
  __shared__ int hist[Sz][Tz];
  const int t = threadIdx.x;

  if (t < Tz) score[t] = start[t] + E[t];
  __syncthreads();
  for (int s = 1; s < Sz; ++s) {
    float m = (float)mask[b * Sz + s];
    if (t < Tz) {
      float best = -3.402823e38f;
      int bi = 0;
      for (int i = 0; i < Tz; ++i) {
        float c = score[i] + trans[i * Tz + t];
        if (c > best) { best = c; bi = i; }
      }
      hist[s][t] = bi;
      ns[t] = (m > 0.0f) ? (best + E[s * Tz + t]) : score[t];
    }
    __syncthreads();
    if (t < Tz) score[t] = ns[t];
    __syncthreads();
  }
  if (t == 0) {
    float best = -3.402823e38f;
    int last = 0;
    for (int i = 0; i < Tz; ++i) {
      float c = score[i] + endv[i];
      if (c > best) { best = c; last = i; }
    }
    int tg[Sz];
    tg[Sz - 1] = last;
    for (int s = Sz - 2; s >= 0; --s) {
      int m = mask[b * Sz + s + 1];
      tg[s] = (m > 0) ? hist[s + 1][tg[s + 1]] : tg[s + 1];
    }
    for (int s = 0; s < Sz; ++s) pred[b * Sz + s] = (float)tg[s];
  }
}

// ---------------------------------------------------------------------------
// Host orchestration
// ---------------------------------------------------------------------------
extern "C" void kernel_launch(void* const* d_in, const int* in_sizes, int n_in,
                              void* d_out, int out_size, void* d_ws, size_t ws_size,
                              hipStream_t stream) {
  (void)in_sizes; (void)n_in; (void)out_size; (void)ws_size;

  const int*   ids    = (const int*)d_in[0];
  const int*   amask  = (const int*)d_in[1];
  const int*   label  = (const int*)d_in[2];
  const float* wemb   = (const float*)d_in[3];
  const float* pemb   = (const float*)d_in[4];
  const float* temb   = (const float*)d_in[5];
  const float* eg     = (const float*)d_in[6];
  const float* eb     = (const float*)d_in[7];
  const float* Wqkv   = (const float*)d_in[8];
  const float* bqkv   = (const float*)d_in[9];
  const float* Wo     = (const float*)d_in[10];
  const float* bo     = (const float*)d_in[11];
  const float* ln1g   = (const float*)d_in[12];
  const float* ln1b   = (const float*)d_in[13];
  const float* W1     = (const float*)d_in[14];
  const float* b1     = (const float*)d_in[15];
  const float* W2     = (const float*)d_in[16];
  const float* b2     = (const float*)d_in[17];
  const float* ln2g   = (const float*)d_in[18];
  const float* ln2b   = (const float*)d_in[19];
  const float* fcw    = (const float*)d_in[20];
  const float* fcb    = (const float*)d_in[21];
  const float* cstart = (const float*)d_in[22];
  const float* cend   = (const float*)d_in[23];
  const float* ctrans = (const float*)d_in[24];

  float* out = (float*)d_out;

  // workspace carve-up
  size_t off = 0;
  auto carve = [&](size_t bytes) -> void* {
    void* p = (char*)d_ws + off;
    off += (bytes + 255) & ~(size_t)255;
    return p;
  };
  float*  x      = (float*)carve((size_t)NBS * Hz * 4);
  __bf16* xb     = (__bf16*)carve((size_t)NBS * Hz * 2);
  __bf16* qkvb   = (__bf16*)carve((size_t)NBS * QKVN * 2);
  __bf16* attnb  = (__bf16*)carve((size_t)NBS * Hz * 2);
  float*  tmp    = (float*)carve((size_t)NBS * Hz * 4);
  __bf16* hb     = (__bf16*)carve((size_t)NBS * Fz * 2);
  __bf16* wqkv_t = (__bf16*)carve((size_t)Hz * QKVN * 2);   // [2304 x 768]
  __bf16* wo_t   = (__bf16*)carve((size_t)Hz * Hz * 2);     // [768 x 768]
  __bf16* w1_t   = (__bf16*)carve((size_t)Hz * Fz * 2);     // [3072 x 768]
  __bf16* w2_t   = (__bf16*)carve((size_t)Fz * Hz * 2);     // [768 x 3072]
  float*  logits = (float*)carve((size_t)NBS * Tz * 4);
  float*  llh    = (float*)carve((size_t)Bz * 4);

  // embeddings + LN
  embed_ln_kernel<<<NBS, 256, 0, stream>>>(ids, wemb, pemb, temb, eg, eb, x, xb);

  for (int l = 0; l < Lz; ++l) {
    const long nqkv = (long)Hz * QKVN;
    const long no   = (long)Hz * Hz;
    const long n1   = (long)Hz * Fz;
    // convert + transpose weights: src [K,N] f32 -> dst [N,K] bf16
    cvt_transpose_kernel<<<dim3(QKVN / 32, Hz / 32), 256, 0, stream>>>(Wqkv + (long)l * nqkv, wqkv_t, Hz, QKVN);
    cvt_transpose_kernel<<<dim3(Hz / 32,   Hz / 32), 256, 0, stream>>>(Wo   + (long)l * no,   wo_t,   Hz, Hz);
    cvt_transpose_kernel<<<dim3(Fz / 32,   Hz / 32), 256, 0, stream>>>(W1   + (long)l * n1,   w1_t,   Hz, Fz);
    cvt_transpose_kernel<<<dim3(Hz / 32,   Fz / 32), 256, 0, stream>>>(W2   + (long)l * n1,   w2_t,   Fz, Hz);

    // QKV projection: [16384,768] @ [768,2304]
    gemm_bf16_kernel<0><<<dim3(QKVN / BN, NBS / BM), 256, 0, stream>>>(
        xb, wqkv_t, bqkv + (long)l * QKVN, nullptr, qkvb, NBS, QKVN, Hz);

    // fused attention per (b, h)
    attention_kernel<<<Bz * NHz, 256, 0, stream>>>(qkvb, amask, attnb);

    // output projection: [16384,768] @ [768,768]
    gemm_bf16_kernel<0><<<dim3(Hz / BN, NBS / BM), 256, 0, stream>>>(
        attnb, wo_t, bo + (long)l * Hz, tmp, nullptr, NBS, Hz, Hz);

    residual_ln_kernel<<<NBS, 256, 0, stream>>>(x, tmp, ln1g + (long)l * Hz, ln1b + (long)l * Hz, xb);

    // FFN1 + gelu: [16384,768] @ [768,3072]
    gemm_bf16_kernel<1><<<dim3(Fz / BN, NBS / BM), 256, 0, stream>>>(
        xb, w1_t, b1 + (long)l * Fz, nullptr, hb, NBS, Fz, Hz);

    // FFN2: [16384,3072] @ [3072,768]
    gemm_bf16_kernel<0><<<dim3(Hz / BN, NBS / BM), 256, 0, stream>>>(
        hb, w2_t, b2 + (long)l * Hz, tmp, nullptr, NBS, Hz, Fz);

    residual_ln_kernel<<<NBS, 256, 0, stream>>>(x, tmp, ln2g + (long)l * Hz, ln2b + (long)l * Hz, xb);
  }

  // tag logits
  logits_kernel<<<NBS, 64, 0, stream>>>(x, fcw, fcb, logits);

  // CRF loss
  crf_llh_kernel<<<Bz, 32, 0, stream>>>(logits, label, amask, cstart, cend, ctrans, llh);
  loss_kernel<<<1, 256, 0, stream>>>(llh, out + NBS);

  // Viterbi decode -> pred
  viterbi_kernel<<<Bz, 32, 0, stream>>>(logits, amask, cstart, cend, ctrans, out);
}
